// Bidirectional_LSTM_Decoder_19078244729398
// MI455X (gfx1250) — matrix operand order repaired and checked by reading, not compiled, not run on hardware
//
#include <hip/hip_runtime.h>
#include <hip/hip_bf16.h>

#define Bsz 32
#define Ssz 2048
#define Dsz 256
#define Hsz 256
#define G4H 1024
#define WELEM (G4H * Dsz)   // 262144 elements per weight matrix

typedef __attribute__((ext_vector_type(16))) _Float16 v16h;
typedef __attribute__((ext_vector_type(8)))  _Float16 v8h;
typedef __attribute__((ext_vector_type(8)))  float    v8f;
typedef __attribute__((ext_vector_type(4)))  int      v4i;

#define AS1 __attribute__((address_space(1)))
#define AS3 __attribute__((address_space(3)))

#if defined(__gfx1250__) && __has_builtin(__builtin_amdgcn_global_load_async_to_lds_b128)
#define ASYNC_OK 1
#else
#define ASYNC_OK 0
#endif

__device__ __forceinline__ v16h cat8(v8h lo, v8h hi) {
    return __builtin_shufflevector(lo, hi, 0,1,2,3,4,5,6,7,8,9,10,11,12,13,14,15);
}

__device__ __forceinline__ float sigm(float v) { return 1.0f / (1.0f + __expf(-v)); }

// acc[4] += A(16 rows starting m0, K=256, from f16 LDS) @ Wn^T
// Wn = weight matrix pre-offset to this wave's 64-row slab (row-major [n][256]).
// A fragment: lane<16 -> row lane, K k0+0..15 ; lane>=16 -> K k0+16..31.
// B fragment: lane = column (lane&15), K half selected by lane bit 4 (0..15 / 16..31).
__device__ __forceinline__ void mm_accum(v8f acc[4],
                                         const _Float16* __restrict__ Ah,
                                         const _Float16* __restrict__ Wn,
                                         int m0, int lane) {
    const _Float16* ap = Ah + (m0 + (lane & 15)) * Dsz + ((lane & 16) ? 16 : 0);
    const _Float16* bp = Wn + (lane & 15) * Dsz + ((lane & 16) ? 16 : 0);
#pragma unroll 2
    for (int kk = 0; kk < 8; ++kk) {
        const int k0 = kk * 32;
        v16h a  = cat8(*(const v8h*)(ap + k0),            *(const v8h*)(ap + k0 + 8));
        v16h b0 = cat8(*(const v8h*)(bp + k0),            *(const v8h*)(bp + k0 + 8));
        v16h b1 = cat8(*(const v8h*)(bp + 16 * Dsz + k0), *(const v8h*)(bp + 16 * Dsz + k0 + 8));
        v16h b2 = cat8(*(const v8h*)(bp + 32 * Dsz + k0), *(const v8h*)(bp + 32 * Dsz + k0 + 8));
        v16h b3 = cat8(*(const v8h*)(bp + 48 * Dsz + k0), *(const v8h*)(bp + 48 * Dsz + k0 + 8));
        acc[0] = __builtin_amdgcn_wmma_f32_16x16x32_f16(false, a, false, b0, (short)0, acc[0], false, false);
        acc[1] = __builtin_amdgcn_wmma_f32_16x16x32_f16(false, a, false, b1, (short)0, acc[1], false, false);
        acc[2] = __builtin_amdgcn_wmma_f32_16x16x32_f16(false, a, false, b2, (short)0, acc[2], false, false);
        acc[3] = __builtin_amdgcn_wmma_f32_16x16x32_f16(false, a, false, b3, (short)0, acc[3], false, false);
    }
}

// Stage one timestep's x tile (32 rows x 256 halfs = 16KB) into LDS.
// One b128 chunk (8 halfs) per thread; async when the CDNA5 builtin exists.
__device__ __forceinline__ void stage_x(const _Float16* __restrict__ xh,
                                        _Float16* __restrict__ dst, int t, int tid) {
    const int row   = tid >> 5;  // batch 0..31
    const int chunk = tid & 31;  // 8-half chunk 0..31
    const _Float16* g = xh + (size_t)row * (Ssz * Dsz) + (size_t)t * Dsz + chunk * 8;
    _Float16* l = dst + row * Dsz + chunk * 8;
#if ASYNC_OK
    __builtin_amdgcn_global_load_async_to_lds_b128(
        (AS1 v4i*)g, (AS3 v4i*)l, 0, 0);
#else
    *(v8h*)l = *(const v8h*)g;
#endif
}

__device__ __forceinline__ void wait_async() {
#if ASYNC_OK
#if __has_builtin(__builtin_amdgcn_s_wait_asynccnt)
    __builtin_amdgcn_s_wait_asynccnt(0);
#else
    asm volatile("s_wait_asynccnt 0" ::: "memory");
#endif
#endif
}

// ---------------------------------------------------------------------------
// Kernel 1: pack weights to f16 (feature-reverse layer-0 backward Wih columns),
// fuse biases, and pre-convert x to f16.
// ---------------------------------------------------------------------------
__global__ void lstm_pack_kernel(const float* __restrict__ x,
                                 const float* __restrict__ Wih_f, const float* __restrict__ Whh_f,
                                 const float* __restrict__ bih_f, const float* __restrict__ bhh_f,
                                 const float* __restrict__ Wih_b, const float* __restrict__ Whh_b,
                                 const float* __restrict__ bih_b, const float* __restrict__ bhh_b,
                                 _Float16* __restrict__ Wp, float* __restrict__ biasp,
                                 _Float16* __restrict__ xh) {
    int gid    = blockIdx.x * blockDim.x + threadIdx.x;
    int stride = gridDim.x * blockDim.x;

    for (int id = gid; id < 8 * WELEM; id += stride) {
        int m = id >> 18;          // matrix slot: ((dir*2+layer)*2+which)
        int r = id & (WELEM - 1);
        int n = r >> 8;
        int d = r & 255;
        int dir   = m >> 2;
        int layer = (m >> 1) & 1;
        int which = m & 1;         // 0 = ih, 1 = hh
        const float* W = dir ? (which ? Whh_b : Wih_b)
                             : (which ? Whh_f : Wih_f);
        int dsrc = (dir == 1 && layer == 0 && which == 0) ? (255 - d) : d;
        Wp[id] = (_Float16)W[layer * WELEM + n * Dsz + dsrc];
    }
    for (int id = gid; id < 4 * G4H; id += stride) {
        int pair = id >> 10;       // dir*2+layer
        int n    = id & 1023;
        int dir = pair >> 1, layer = pair & 1;
        const float* bi = dir ? bih_b : bih_f;
        const float* bh = dir ? bhh_b : bhh_f;
        biasp[id] = bi[layer * G4H + n] + bh[layer * G4H + n];
    }
    for (int id = gid; id < Bsz * Ssz * Dsz; id += stride) {
        xh[id] = (_Float16)x[id];
    }
}

// ---------------------------------------------------------------------------
// Kernel 2: persistent recurrent kernel, one workgroup per direction.
// 32 waves: wave w -> M tile (w&1)*16, N slab (w>>1)*64 (4 x 16 tiles).
// x tile for step t+1 async-prefetched to LDS while step t computes.
// ---------------------------------------------------------------------------
__global__ void lstm_recurrent_kernel(const _Float16* __restrict__ xh,
                                      const float* __restrict__ enc_h,
                                      const float* __restrict__ enc_c,
                                      const _Float16* __restrict__ Wp,
                                      const float* __restrict__ biasp,
                                      float* __restrict__ out) {
    extern __shared__ char smem[];
    float*    G  = (float*)smem;                          // 32 x 1024 f32  (128 KB)
    _Float16* xl = (_Float16*)(smem + Bsz * G4H * 4);     // 2 x 32 x 256 f16 (32 KB)
    _Float16* h0 = xl + 2 * Bsz * Hsz;                    // 32 x 256 f16
    _Float16* h1 = h0 + Bsz * Hsz;                        // 32 x 256 f16
    float*    c0 = (float*)(h1 + Bsz * Hsz);              // 32 x 256 f32
    float*    c1 = c0 + Bsz * Hsz;                        // 32 x 256 f32

    const int dir   = blockIdx.x;
    const int tid   = threadIdx.x;
    const int lane  = tid & 31;
    const int wave  = tid >> 5;
    const int m0    = (wave & 1) * 16;
    const int nBase = (wave >> 1) * 64;

    // wave-local weight slabs (64 gate rows each)
    const _Float16* Wih0 = Wp + ((dir * 2 + 0) * 2 + 0) * WELEM + nBase * Dsz;
    const _Float16* Whh0 = Wp + ((dir * 2 + 0) * 2 + 1) * WELEM + nBase * Dsz;
    const _Float16* Wih1 = Wp + ((dir * 2 + 1) * 2 + 0) * WELEM + nBase * Dsz;
    const _Float16* Whh1 = Wp + ((dir * 2 + 1) * 2 + 1) * WELEM + nBase * Dsz;
    const float* bias0 = biasp + (dir * 2 + 0) * G4H;
    const float* bias1 = biasp + (dir * 2 + 1) * G4H;

    // stage x tile for t=0, init h/c from encoder state
    stage_x(xh, xl, 0, tid);
    for (int e = tid; e < Bsz * Hsz; e += blockDim.x) {
        int b = e >> 8, j = e & 255;
        float hv = enc_h[b * 512 + dir * Hsz + j];
        float cv = enc_c[b * 512 + dir * Hsz + j];
        h0[e] = (_Float16)hv;
        h1[e] = (_Float16)hv;
        c0[e] = cv;
        c1[e] = cv;
    }
    wait_async();
    __syncthreads();

    for (int t = 0; t < Ssz; ++t) {
        const _Float16* xbuf = xl + (t & 1) * (Bsz * Hsz);

        // ================= layer 0 =================
        v8f acc[4];
#pragma unroll
        for (int j = 0; j < 4; ++j) {
            float bv = bias0[nBase + 16 * j + (lane & 15)];
#pragma unroll
            for (int i = 0; i < 8; ++i) acc[j][i] = bv;
        }
        mm_accum(acc, xbuf, Wih0, m0, lane);
        mm_accum(acc, h0,   Whh0, m0, lane);
#pragma unroll
        for (int j = 0; j < 4; ++j) {
            int col = nBase + 16 * j + (lane & 15);
#pragma unroll
            for (int i = 0; i < 8; ++i)
                G[(m0 + i + ((lane & 16) ? 8 : 0)) * G4H + col] = acc[j][i];
        }
        __syncthreads();

        // prefetch next step's x tile into the other buffer (its last reader
        // finished during step t-1, before this step began)
        if (t + 1 < Ssz)
            stage_x(xh, xl + ((t + 1) & 1) * (Bsz * Hsz), t + 1, tid);

        // elementwise gate update, layer 0
        for (int r = 0; r < 8; ++r) {
            int e = r * 1024 + tid;
            int b = e >> 8, j = e & 255;
            const float* g = G + b * G4H;
            float gi = g[j], gf = g[256 + j], gg = g[512 + j], go = g[768 + j];
            float cn = sigm(gf) * c0[e] + sigm(gi) * tanhf(gg);
            float hn = sigm(go) * tanhf(cn);
            c0[e] = cn;
            h0[e] = (_Float16)hn;
        }
        __syncthreads();

        // ================= layer 1 =================
#pragma unroll
        for (int j = 0; j < 4; ++j) {
            float bv = bias1[nBase + 16 * j + (lane & 15)];
#pragma unroll
            for (int i = 0; i < 8; ++i) acc[j][i] = bv;
        }
        mm_accum(acc, h0, Wih1, m0, lane);
        mm_accum(acc, h1, Whh1, m0, lane);
#pragma unroll
        for (int j = 0; j < 4; ++j) {
            int col = nBase + 16 * j + (lane & 15);
#pragma unroll
            for (int i = 0; i < 8; ++i)
                G[(m0 + i + ((lane & 16) ? 8 : 0)) * G4H + col] = acc[j][i];
        }
        __syncthreads();

        // elementwise gate update, layer 1, + output write
        for (int r = 0; r < 8; ++r) {
            int e = r * 1024 + tid;
            int b = e >> 8, j = e & 255;
            const float* g = G + b * G4H;
            float gi = g[j], gf = g[256 + j], gg = g[512 + j], go = g[768 + j];
            float cn = sigm(gf) * c1[e] + sigm(gi) * tanhf(gg);
            float hn = sigm(go) * tanhf(cn);
            c1[e] = cn;
            h1[e] = (_Float16)hn;
            out[(size_t)b * Ssz * 512 + (size_t)t * 512 + dir * Hsz + j] = hn;
        }
        wait_async();   // this wave's prefetch of t+1 is done
        __syncthreads();
    }

    // final states: d_out = [out | h_last | c_last]
    float* hlast = out + (size_t)Bsz * Ssz * 512;
    float* clast = hlast + Bsz * 512;
    for (int e = tid; e < Bsz * Hsz; e += blockDim.x) {
        int b = e >> 8, j = e & 255;
        hlast[b * 512 + dir * Hsz + j] = (float)h1[e];
        clast[b * 512 + dir * Hsz + j] = c1[e];
    }
}

extern "C" void kernel_launch(void* const* d_in, const int* in_sizes, int n_in,
                              void* d_out, int out_size, void* d_ws, size_t ws_size,
                              hipStream_t stream) {
    const float* x     = (const float*)d_in[0];
    const float* enc_h = (const float*)d_in[1];
    const float* enc_c = (const float*)d_in[2];
    const float* Wih_f = (const float*)d_in[3];
    const float* Whh_f = (const float*)d_in[4];
    const float* bih_f = (const float*)d_in[5];
    const float* bhh_f = (const float*)d_in[6];
    const float* Wih_b = (const float*)d_in[7];
    const float* Whh_b = (const float*)d_in[8];
    const float* bih_b = (const float*)d_in[9];
    const float* bhh_b = (const float*)d_in[10];

    // workspace layout: Wp (4 MB f16) | biasp (16 KB f32) | xh (32 MB f16)
    _Float16* Wp    = (_Float16*)d_ws;
    float*    biasp = (float*)((char*)d_ws + (size_t)8 * WELEM * 2);
    _Float16* xh    = (_Float16*)((char*)d_ws + (size_t)8 * WELEM * 2 + 4 * G4H * 4);

    lstm_pack_kernel<<<8192, 256, 0, stream>>>(x, Wih_f, Whh_f, bih_f, bhh_f,
                                               Wih_b, Whh_b, bih_b, bhh_b,
                                               Wp, biasp, xh);

    size_t lds_bytes = (size_t)Bsz * G4H * 4       // G       128 KB
                     + (size_t)2 * Bsz * Hsz * 2   // x dbuf   32 KB
                     + (size_t)2 * Bsz * Hsz * 2   // h0,h1    32 KB
                     + (size_t)2 * Bsz * Hsz * 4;  // c0,c1    64 KB
    lstm_recurrent_kernel<<<2, 1024, lds_bytes, stream>>>(xh, enc_h, enc_c,
                                                          Wp, biasp, (float*)d_out);
}
